// GLASSv2Layer_83288005804431
// MI455X (gfx1250) — matrix operand order repaired and checked by reading, not compiled, not run on hardware
//
#include <hip/hip_runtime.h>
#include <hip/hip_bf16.h>

#define DIMC 128
#define HIDC 256
#define NHC 8
#define HDC 32
#define FFNC 256
#define BSZ 2
#define HIMG 256
#define WIMG 256
#define NPOS (HIMG * WIMG)
#define TTOK (BSZ * NPOS)

typedef __attribute__((ext_vector_type(16))) __bf16 v16bf;
typedef __attribute__((ext_vector_type(8))) float v8f;

union FragBF {
  v16bf v;
  uint4 u[2];
};

__device__ __forceinline__ unsigned short f2bf(float f) {
  union { float f; unsigned u; } cv; cv.f = f;
  unsigned r = cv.u + 0x7FFFu + ((cv.u >> 16) & 1u);
  return (unsigned short)(r >> 16);
}

__device__ __forceinline__ float gelu_exact(float x) {
  return 0.5f * x * (1.0f + erff(x * 0.7071067811865476f));
}

__device__ __forceinline__ float wave_sum(float v) {
#pragma unroll
  for (int off = 16; off > 0; off >>= 1) v += __shfl_xor(v, off, 32);
  return v;
}

// ---------------------------------------------------------------------------
// f32 -> bf16 conversion (weights)
// ---------------------------------------------------------------------------
__global__ void cvt_bf16_kernel(const float* __restrict__ in,
                                unsigned short* __restrict__ out, int n) {
  int i = blockIdx.x * 256 + threadIdx.x;
  if (i < n) out[i] = f2bf(in[i]);
}

// ---------------------------------------------------------------------------
// LayerNorm over C (128 or 256), one wave per token, bf16 output
// ---------------------------------------------------------------------------
__global__ __launch_bounds__(256)
void ln_bf16_kernel(const float* __restrict__ in, const float* __restrict__ g,
                    const float* __restrict__ b, unsigned short* __restrict__ outB,
                    int C) {
  const int wave = threadIdx.x >> 5;
  const int lane = threadIdx.x & 31;
  const size_t t = (size_t)blockIdx.x * 8 + wave;
  const float* xp = in + t * C;
  const int E = C >> 5;
  float vals[8];
  float s = 0.0f;
  for (int e = 0; e < E; ++e) {
    vals[e] = xp[e * 32 + lane];
    s += vals[e];
  }
  const float mean = wave_sum(s) / (float)C;
  float s2 = 0.0f;
  for (int e = 0; e < E; ++e) {
    float d = vals[e] - mean;
    s2 += d * d;
  }
  const float inv = rsqrtf(wave_sum(s2) / (float)C + 1e-5f);
  unsigned short* op = outB + t * C;
  for (int e = 0; e < E; ++e) {
    int c = e * 32 + lane;
    op[c] = f2bf((vals[e] - mean) * inv * g[c] + b[c]);
  }
}

// ---------------------------------------------------------------------------
// bf16 WMMA GEMM: out[M x Nout] = A[M x K] * W[Nout x K]^T (+bias,+resid,act)
// block = 8 waves -> 64M x 128N tile; wave = 16M x 64N (4 accumulators)
// ---------------------------------------------------------------------------
__global__ __launch_bounds__(256)
void gemm_bf16_kernel(const unsigned short* __restrict__ A,
                      const unsigned short* __restrict__ Wt,
                      const float* __restrict__ bias,
                      const float* __restrict__ resid,
                      float* __restrict__ outF,
                      unsigned short* __restrict__ outB,
                      int M, int K, int Nout, int act) {
  const int wave = threadIdx.x >> 5;
  const int lane = threadIdx.x & 31;
  const int hi   = lane >> 4;      // half-wave select
  const int ln16 = lane & 15;
  const int mrow = (wave & 3) * 16;
  const int ncol = (wave >> 2) * 64;
  const int Mbase = blockIdx.x * 64;
  const int Nbase = blockIdx.y * 128;
  const int row = Mbase + mrow + ln16;

  v8f acc[4] = {};

  const unsigned short* Arow = A + (size_t)row * K;
  for (int kk = 0; kk < K; kk += 32) {
    // A fragment: 16x32 bf16. Per ISA: lane half 'hi' owns K = kk+8*hi+[0..7]
    // (VGPR0-3) and K = kk+16+8*hi+[0..7] (VGPR4-7).
    FragBF a;
    a.u[0] = *(const uint4*)(Arow + kk + hi * 8);
    a.u[1] = *(const uint4*)(Arow + kk + 16 + hi * 8);
    if (kk + 64 < K) __builtin_prefetch(Arow + kk + 64, 0, 0);
#pragma unroll
    for (int j = 0; j < 4; ++j) {
      // B fragment: 32x16 bf16; lane = column n, K = kk+8*hi+[0..7] and kk+16+8*hi+[0..7].
      const int n = Nbase + ncol + j * 16 + ln16;
      const unsigned short* Wr = Wt + (size_t)n * K;
      FragBF bm;
      bm.u[0] = *(const uint4*)(Wr + kk + hi * 8);
      bm.u[1] = *(const uint4*)(Wr + kk + 16 + hi * 8);
      acc[j] = __builtin_amdgcn_wmma_f32_16x16x32_bf16(
          false, a.v, false, bm.v, (short)0, acc[j], false, false);
    }
  }

#pragma unroll
  for (int j = 0; j < 4; ++j) {
    const int n = Nbase + ncol + j * 16 + ln16;
    const float bi = bias ? bias[n] : 0.0f;
#pragma unroll
    for (int r = 0; r < 8; ++r) {
      const int m = Mbase + mrow + r + 8 * hi;   // C layout: M = r + 8*hi
      const size_t o = (size_t)m * Nout + n;
      float v = acc[j][r] + bi;
      if (resid) v += resid[o];
      if (act) v = gelu_exact(v);
      if (outF) outF[o] = v;
      if (outB) outB[o] = f2bf(v);
    }
  }
}

// ---------------------------------------------------------------------------
// LePE: depthwise 3x3 over v (qkv slice, stride 768), vnew = v + conv
// ---------------------------------------------------------------------------
__global__ __launch_bounds__(256)
void lepe_kernel(const float* __restrict__ qkv, const float* __restrict__ w,
                 const float* __restrict__ bias, float* __restrict__ vnew) {
  const int t = blockIdx.x;
  const int c = threadIdx.x;
  const int b = t >> 16;           // / NPOS
  const int n = t & (NPOS - 1);
  const int y = n >> 8;            // / WIMG
  const int x = n & 255;
  float acc = bias[c];
  const float* wc = w + c * 9;
#pragma unroll
  for (int ky = 0; ky < 3; ++ky) {
    const int yy = y + ky - 1;
    if (yy < 0 || yy >= HIMG) continue;
#pragma unroll
    for (int kx = 0; kx < 3; ++kx) {
      const int xx = x + kx - 1;
      if (xx < 0 || xx >= WIMG) continue;
      const size_t src = (size_t)(b * NPOS + yy * WIMG + xx) * 768 + 512 + c;
      acc += wc[ky * 3 + kx] * qkv[src];
    }
  }
  vnew[(size_t)t * HIDC + c] = qkv[(size_t)t * 768 + 512 + c] + acc;
}

// ---------------------------------------------------------------------------
// RoPE + elu+1 (phi) applied in-place on q and k slices of qkv.
// thread p<128 -> q, p>=128 -> k; handles paired dims (d, d+16) of one head.
// ---------------------------------------------------------------------------
__global__ __launch_bounds__(256)
void rope_phi_kernel(float* __restrict__ qkv) {
  const int t = blockIdx.x;
  const int p = threadIdx.x;
  const int isK = p >> 7;
  const int pp = p & 127;
  const int h = pp >> 4;
  const int d = pp & 15;
  const int n = t & (NPOS - 1);
  const int y = n >> 8;
  const int x = n & 255;
  float* base = qkv + (size_t)t * 768 + isK * 256 + h * HDC;
  const float xa = base[d];
  const float xb = base[d + 16];
  // freq[0:8]=th, [8:16]=th, [16:24]=tw, [24:32]=tw ; j = d % 8
  const float invf = __expf(-(float)(d & 7) * 1.1512925464970231f); // ln(1e4)/8
  float sa, ca, sb, cb;
  __sincosf((float)y * invf, &sa, &ca);
  __sincosf((float)x * invf, &sb, &cb);
  const float oa = xa * ca - xb * sa;
  const float ob = xb * cb + xa * sb;
  base[d]      = (oa > 0.0f) ? oa + 1.0f : __expf(oa);
  base[d + 16] = (ob > 0.0f) ? ob + 1.0f : __expf(ob);
}

// ---------------------------------------------------------------------------
// kv accumulation: kv[b][h][32][32] = sum_n phi_k ⊗ vnew ; ksum[b][h][32]
// grid = (64 chunks, 16 bh); each block does 1024 tokens (8 LDS passes of 128)
// ---------------------------------------------------------------------------
__global__ __launch_bounds__(256)
void kv_reduce_kernel(const float* __restrict__ qkv, const float* __restrict__ vnew,
                      float* __restrict__ kvbuf) {
  __shared__ float sk[128 * 32];
  __shared__ float sv[128 * 32];
  const int bh = blockIdx.y;
  const int b = bh >> 3;
  const int h = bh & 7;
  const int tid = threadIdx.x;
  const int myd = tid >> 3;
  const int e0 = (tid & 7) * 4;
  const size_t tok0 = (size_t)b * NPOS + (size_t)blockIdx.x * 1024;
  float acc[4] = {0.0f, 0.0f, 0.0f, 0.0f};
  float ks = 0.0f;

  for (int sub = 0; sub < 8; ++sub) {
    for (int i = tid; i < 4096; i += 256) {
      const int tok = i >> 5;
      const int d = i & 31;
      const size_t gt = tok0 + sub * 128 + tok;
      sk[i] = qkv[gt * 768 + 256 + h * HDC + d];
      sv[i] = vnew[gt * HIDC + h * HDC + d];
    }
    __syncthreads();
    for (int tok = 0; tok < 128; ++tok) {
      const float kd = sk[tok * 32 + myd];
      const float* vv = &sv[tok * 32 + e0];
      acc[0] += kd * vv[0];
      acc[1] += kd * vv[1];
      acc[2] += kd * vv[2];
      acc[3] += kd * vv[3];
      if (tid < 32) ks += sk[tok * 32 + tid];
    }
    __syncthreads();
  }

  float* kvb = kvbuf + (size_t)bh * 1056;
#pragma unroll
  for (int j = 0; j < 4; ++j) atomicAdd(&kvb[myd * 32 + e0 + j], acc[j]);
  if (tid < 32) atomicAdd(&kvb[1024 + tid], ks);
}

// ---------------------------------------------------------------------------
// out = (phi_q @ kv) / max(phi_q . ksum, 1e-6), written dense T x 256
// ---------------------------------------------------------------------------
__global__ __launch_bounds__(256)
void attn_out_kernel(const float* __restrict__ qkv, const float* __restrict__ kvbuf,
                     float* __restrict__ out) {
  const int t = blockIdx.x;
  const int c = threadIdx.x;
  const int h = c >> 5;
  const int e = c & 31;
  const int b = t >> 16;
  const float* kvb = kvbuf + (size_t)(b * NHC + h) * 1056;
  const float* qp = qkv + (size_t)t * 768 + h * HDC;
  float num = 0.0f, den = 0.0f;
#pragma unroll
  for (int d = 0; d < 32; ++d) {
    const float qd = qp[d];
    num += qd * kvb[d * 32 + e];
    den += qd * kvb[1024 + d];
  }
  out[(size_t)t * HIDC + c] = num / fmaxf(den, 1e-6f);
}

// ---------------------------------------------------------------------------
// depthwise 5x5 on h1 (dense T x 256): out_bf16 = h1 + gelu(conv(h1))
// ---------------------------------------------------------------------------
__global__ __launch_bounds__(256)
void dwconv5_kernel(const float* __restrict__ h1, const float* __restrict__ w,
                    const float* __restrict__ bias, unsigned short* __restrict__ outB) {
  const int t = blockIdx.x;
  const int c = threadIdx.x;
  const int b = t >> 16;
  const int n = t & (NPOS - 1);
  const int y = n >> 8;
  const int x = n & 255;
  float acc = bias[c];
  const float* wc = w + c * 25;
#pragma unroll
  for (int ky = 0; ky < 5; ++ky) {
    const int yy = y + ky - 2;
    if (yy < 0 || yy >= HIMG) continue;
#pragma unroll
    for (int kx = 0; kx < 5; ++kx) {
      const int xx = x + kx - 2;
      if (xx < 0 || xx >= WIMG) continue;
      acc += wc[ky * 5 + kx] * h1[(size_t)(b * NPOS + yy * WIMG + xx) * FFNC + c];
    }
  }
  const float s = h1[(size_t)t * FFNC + c] + gelu_exact(acc);
  outB[(size_t)t * FFNC + c] = f2bf(s);
}

// ---------------------------------------------------------------------------
// host launch
// ---------------------------------------------------------------------------
extern "C" void kernel_launch(void* const* d_in, const int* in_sizes, int n_in,
                              void* d_out, int out_size, void* d_ws, size_t ws_size,
                              hipStream_t stream) {
  (void)in_sizes; (void)n_in; (void)out_size; (void)ws_size;

  const float* x          = (const float*)d_in[0];
  const float* in_proj_w  = (const float*)d_in[3];
  const float* in_proj_b  = (const float*)d_in[4];
  const float* qkv_w      = (const float*)d_in[5];
  const float* qkv_b      = (const float*)d_in[6];
  const float* lepe_w     = (const float*)d_in[7];
  const float* lepe_b     = (const float*)d_in[8];
  const float* onorm_g    = (const float*)d_in[9];
  const float* onorm_b    = (const float*)d_in[10];
  const float* out_proj_w = (const float*)d_in[11];
  const float* out_proj_b = (const float*)d_in[12];
  const float* norm1_g    = (const float*)d_in[13];
  const float* norm1_b    = (const float*)d_in[14];
  const float* norm2_g    = (const float*)d_in[15];
  const float* norm2_b    = (const float*)d_in[16];
  const float* fc1_w      = (const float*)d_in[17];
  const float* fc1_b      = (const float*)d_in[18];
  const float* dw_w       = (const float*)d_in[19];
  const float* dw_b       = (const float*)d_in[20];
  const float* fc2_w      = (const float*)d_in[21];
  const float* fc2_b      = (const float*)d_in[22];
  float* out              = (float*)d_out;

  char* ws = (char*)d_ws;
  // bf16 weights (front of workspace)
  unsigned short* w_in  = (unsigned short*)ws;                 // 256*128
  unsigned short* w_qkv = w_in + 32768;                        // 768*256
  unsigned short* w_out = w_qkv + 196608;                      // 128*256
  unsigned short* w_fc1 = w_out + 32768;                       // 256*128
  unsigned short* w_fc2 = w_fc1 + 32768;                       // 128*256
  float* kvbuf = (float*)(ws + 720896);                        // 16 * (1024+32)
  size_t off = (size_t)1 << 20;
  unsigned short* bufA128 = (unsigned short*)(ws + off); off += (size_t)TTOK * 128 * 2; // ln1 out / ln2 out
  unsigned short* bufA256 = (unsigned short*)(ws + off); off += (size_t)TTOK * 256 * 2; // xs / ln_attn / ffn sum
  float* bufQKV = (float*)(ws + off); off += (size_t)TTOK * 768 * 4;                    // qkv, later h1
  float* bufV   = (float*)(ws + off); off += (size_t)TTOK * 256 * 4;                    // v+lepe, later attn out
  float* bufX1  = (float*)(ws + off); off += (size_t)TTOK * 128 * 4;                    // x after attention

  // --- weight conversion (deterministic, every launch) ---
  cvt_bf16_kernel<<<(32768 + 255) / 256, 256, 0, stream>>>(in_proj_w, w_in, 32768);
  cvt_bf16_kernel<<<(196608 + 255) / 256, 256, 0, stream>>>(qkv_w, w_qkv, 196608);
  cvt_bf16_kernel<<<(32768 + 255) / 256, 256, 0, stream>>>(out_proj_w, w_out, 32768);
  cvt_bf16_kernel<<<(32768 + 255) / 256, 256, 0, stream>>>(fc1_w, w_fc1, 32768);
  cvt_bf16_kernel<<<(32768 + 255) / 256, 256, 0, stream>>>(fc2_w, w_fc2, 32768);

  const dim3 blk(256);

  // 1) y = LN1(x) -> bf16
  ln_bf16_kernel<<<TTOK / 8, blk, 0, stream>>>(x, norm1_g, norm1_b, bufA128, 128);
  // 2) xs = y @ in_proj^T + b -> bf16
  gemm_bf16_kernel<<<dim3(TTOK / 64, 2), blk, 0, stream>>>(
      bufA128, w_in, in_proj_b, nullptr, nullptr, bufA256, TTOK, 128, 256, 0);
  // 3) qkv = xs @ qkv^T + b -> f32 (T x 768)
  gemm_bf16_kernel<<<dim3(TTOK / 64, 6), blk, 0, stream>>>(
      bufA256, w_qkv, qkv_b, nullptr, bufQKV, nullptr, TTOK, 256, 768, 0);
  // 4) vnew = v + lepe(v)
  lepe_kernel<<<TTOK, blk, 0, stream>>>(bufQKV, lepe_w, lepe_b, bufV);
  // 5) phi(rope(q)), phi(rope(k)) in place
  rope_phi_kernel<<<TTOK, blk, 0, stream>>>(bufQKV);
  // 6) kv accumulators
  hipMemsetAsync(kvbuf, 0, (size_t)16 * 1056 * 4, stream);
  kv_reduce_kernel<<<dim3(64, 16), blk, 0, stream>>>(bufQKV, bufV, kvbuf);
  // 7) out = (q @ kv) / denom  (reuse bufV)
  attn_out_kernel<<<TTOK, blk, 0, stream>>>(bufQKV, kvbuf, bufV);
  // 8) LN(attn) -> bf16
  ln_bf16_kernel<<<TTOK / 8, blk, 0, stream>>>(bufV, onorm_g, onorm_b, bufA256, 256);
  // 9) x1 = x + LN @ out_proj^T + b
  gemm_bf16_kernel<<<dim3(TTOK / 64, 1), blk, 0, stream>>>(
      bufA256, w_out, out_proj_b, x, bufX1, nullptr, TTOK, 256, 128, 0);
  // 10) y2 = LN2(x1) -> bf16
  ln_bf16_kernel<<<TTOK / 8, blk, 0, stream>>>(bufX1, norm2_g, norm2_b, bufA128, 128);
  // 11) h1 = gelu(y2 @ fc1^T + b) -> f32 dense T x 256 (reuse bufQKV)
  float* h1 = bufQKV;
  gemm_bf16_kernel<<<dim3(TTOK / 64, 2), blk, 0, stream>>>(
      bufA128, w_fc1, fc1_b, nullptr, h1, nullptr, TTOK, 128, 256, 1);
  // 12) hsum = h1 + gelu(dwconv5(h1)) -> bf16
  dwconv5_kernel<<<TTOK, blk, 0, stream>>>(h1, dw_w, dw_b, bufA256);
  // 13) out = x1 + hsum @ fc2^T + b
  gemm_bf16_kernel<<<dim3(TTOK / 64, 1), blk, 0, stream>>>(
      bufA256, w_fc2, fc2_b, bufX1, out, nullptr, TTOK, 256, 128, 0);
}
